// NRIMLP_14654428414498
// MI455X (gfx1250) — compile-verified, matrix operand order checked by reading, not run on hardware
//
#include <hip/hip_runtime.h>
#include <hip/hip_bf16.h>

typedef __bf16 bf16_t;
typedef __attribute__((ext_vector_type(16))) __bf16 v16bf;
typedef __attribute__((ext_vector_type(8)))  float  v8f;
typedef __attribute__((ext_vector_type(4)))  float  f32x4;

#define BATCH   64
#define NATOMS  64
#define EDGES   4032
#define EROWS   (BATCH * EDGES)   // 258048
#define NROWS   (BATCH * NATOMS)  // 4096

// ---------------------------------------------------------------- helpers

__device__ __forceinline__ float elu_f(float x) {
    return x > 0.0f ? x : (__expf(x) - 1.0f);
}

// mode 0: identity ; mode 1: sender node ; mode 2: receiver node
__device__ __forceinline__ int map_row(int mode, int r) {
    if (mode == 0) return r;
    int b  = r / EDGES;
    int e  = r - b * EDGES;
    int i  = e / 63;
    int rr = e - i * 63;
    if (mode == 1) return b * NATOMS + i;
    int j = rr + (rr >= i ? 1 : 0);
    return b * NATOMS + j;
}

union AFrag { f32x4 f[2]; v16bf v; };

// A fragment (16x32 bf16) from row-major LDS tile.
__device__ __forceinline__ v16bf load_frag_rows(const bf16_t* base, int rowStride, int lane) {
    int m = lane & 15, h = lane >> 4;
    const bf16_t* p = base + m * rowStride + h * 8;
    AFrag u;
    u.f[0] = *(const f32x4*)(p);
    u.f[1] = *(const f32x4*)(p + 16);
    return u.v;
}

// B fragment from LDS-staged swizzled slab: 16 contiguous bf16 per lane
__device__ __forceinline__ v16bf load_frag_ws(const bf16_t* slab, int nt, int lane) {
    const bf16_t* p = slab + ((nt * 32) + lane) * 16;
    AFrag u;
    u.f[0] = *(const f32x4*)(p);
    u.f[1] = *(const f32x4*)(p + 8);
    return u.v;
}

// low 32 bits of a generic pointer to an LDS object = workgroup-relative LDS byte address
__device__ __forceinline__ unsigned lds_addr(const void* p) {
    return (unsigned)(unsigned long long)(uintptr_t)p;
}

// CDNA5 async copy: 16B global -> LDS, tracked by ASYNCcnt
__device__ __forceinline__ void async_b128(unsigned ldsOff, const void* g) {
    asm volatile("global_load_async_to_lds_b128 %0, %1, off"
                 :: "v"(ldsOff), "v"((unsigned long long)g)
                 : "memory");
}

__device__ __forceinline__ void wait_async0() {
    asm volatile("s_wait_asynccnt 0x0" ::: "memory");
}

// force all 9 fragments into distinct live registers + scheduling barrier:
// all ds_loads complete (single wait) before the WMMA burst begins
#define PIN_FRAGS(af, bfa)                                                     \
    asm volatile("" : "+v"(af), "+v"(bfa[0]), "+v"(bfa[1]), "+v"(bfa[2]),      \
                      "+v"(bfa[3]), "+v"(bfa[4]), "+v"(bfa[5]), "+v"(bfa[6]),  \
                      "+v"(bfa[7]))

// ---------------------------------------------------------------- GEMM args

struct GemmArgs {
    const bf16_t* seg[3];   // bf16 row-major [rows][256] (BN folded into weights)
    int           rowMode[3];
    const bf16_t* W1;       // swizzled bf16 [K/32][16][32][16], BN scale pre-folded
    const float*  b1;       // folded bias
    const bf16_t* W2;       // swizzled bf16, K=256
    const float*  b2;
    bf16_t*       out;      // pre-BN ELU output, bf16 [rows][256]
    float*        sums;     // [256]
    float*        sumsq;    // [256]
    int           nKB;      // K/32 for layer 1
};

// ---------------------------------------------------------------- fused 2-layer MLP (async double-buffered)

__global__ __launch_bounds__(256) void nri_mlp2layer(GemmArgs a) {
    __shared__ bf16_t As[2][64 * 32];    // activation K-slab, double buffered (8 KB)
    __shared__ bf16_t Ws[2][32 * 256];   // swizzled weight K-slab, double buffered (32 KB)
    __shared__ bf16_t Hs[64 * 256];      // layer-1 output for layer-2 A tiles (32 KB)
    __shared__ float  sSum[256];
    __shared__ float  sSqr[256];

    const int tid  = threadIdx.x;
    const int lane = tid & 31;
    const int wave = tid >> 5;
    const int waveRow  = (wave >> 1) * 16;
    const int ntBase   = (wave & 1) * 8;
    const int rowBlock = blockIdx.x * 64;

    sSum[tid] = 0.0f;
    sSqr[tid] = 0.0f;

    // staging coords: each thread owns one 16B activation chunk and 64B of weight slab
    const int lr = tid >> 2;            // local row 0..63
    const int kq = (tid & 3) * 8;       // K sub-offset 0/8/16/24

    const bf16_t* gbase[3];
#pragma unroll
    for (int s = 0; s < 3; ++s) {
        const bf16_t* sp = a.seg[s];
        gbase[s] = sp ? (sp + (size_t)map_row(a.rowMode[s], rowBlock + lr) * 256 + kq) : nullptr;
    }
    const unsigned asOff[2] = { lds_addr(&As[0][lr * 32 + kq]), lds_addr(&As[1][lr * 32 + kq]) };
    const unsigned wsOff[2] = { lds_addr(&Ws[0][0]) + (unsigned)tid * 64u,
                                lds_addr(&Ws[1][0]) + (unsigned)tid * 64u };

    auto issueA = [&](int kb, int buf) {
        const int kk = kb * 32;
        async_b128(asOff[buf], gbase[kk >> 8] + (kk & 255));
    };
    auto issueW = [&](const bf16_t* W, int kb, int buf) {
        const char* g = (const char*)(W + kb * 8192) + tid * 64;
#pragma unroll
        for (int q = 0; q < 4; ++q) async_b128(wsOff[buf] + q * 16u, g + q * 16);
    };

    const v8f vzero = {0.f,0.f,0.f,0.f,0.f,0.f,0.f,0.f};
    v8f acc[8];
#pragma unroll
    for (int t = 0; t < 8; ++t) acc[t] = vzero;

    // ---------------- layer 1: H = ELU(A * W1 + b1')
    issueA(0, 0);
    issueW(a.W1, 0, 0);
    for (int kb = 0; kb < a.nKB; ++kb) {
        const int cur = kb & 1, nxt = cur ^ 1;
        wait_async0();
        __syncthreads();                       // cur buffers ready for all waves
        if (kb + 1 < a.nKB) {                  // prefetch next slab while computing
            issueA(kb + 1, nxt);
            issueW(a.W1, kb + 1, nxt);
        }
        // batch all fragment loads (one ds clause + single wait), then WMMA burst
        v16bf af = load_frag_rows(&As[cur][waveRow * 32], 32, lane);
        v16bf bf[8];
#pragma unroll
        for (int t = 0; t < 8; ++t) bf[t] = load_frag_ws(&Ws[cur][0], ntBase + t, lane);
        PIN_FRAGS(af, bf);
#pragma unroll
        for (int t = 0; t < 8; ++t) {
            acc[t] = __builtin_amdgcn_wmma_f32_16x16x32_bf16(
                false, af, false, bf[t], (short)0, acc[t], false, false);
        }
    }

    // epilogue 1 -> Hs (bf16)
    const int m0 = waveRow + ((lane >> 4) << 3);
    const int n4 = lane & 15;
#pragma unroll
    for (int t = 0; t < 8; ++t) {
        const int col = (ntBase + t) * 16 + n4;
        const float bias = a.b1[col];
#pragma unroll
        for (int r = 0; r < 8; ++r) {
            float v = elu_f(acc[t][r] + bias);
            Hs[(m0 + r) * 256 + col] = (bf16_t)v;
        }
    }

    // ---------------- layer 2: D = ELU(H * W2 + b2)
#pragma unroll
    for (int t = 0; t < 8; ++t) acc[t] = vzero;
    issueW(a.W2, 0, 0);
#pragma unroll
    for (int kb = 0; kb < 8; ++kb) {
        const int cur = kb & 1, nxt = cur ^ 1;
        wait_async0();
        __syncthreads();                       // also orders epilogue-1 Hs writes (kb==0)
        if (kb < 7) issueW(a.W2, kb + 1, nxt);
        v16bf af = load_frag_rows(&Hs[waveRow * 256 + kb * 32], 256, lane);
        v16bf bf[8];
#pragma unroll
        for (int t = 0; t < 8; ++t) bf[t] = load_frag_ws(&Ws[cur][0], ntBase + t, lane);
        PIN_FRAGS(af, bf);
#pragma unroll
        for (int t = 0; t < 8; ++t) {
            acc[t] = __builtin_amdgcn_wmma_f32_16x16x32_bf16(
                false, af, false, bf[t], (short)0, acc[t], false, false);
        }
    }

    // epilogue 2: ELU -> bf16 out + BN stats (LDS reduce, then 2 global atomics/thread)
#pragma unroll
    for (int t = 0; t < 8; ++t) {
        const int col = (ntBase + t) * 16 + n4;
        const float bias = a.b2[col];
        float s1 = 0.f, s2 = 0.f;
#pragma unroll
        for (int r = 0; r < 8; ++r) {
            float v = elu_f(acc[t][r] + bias);
            a.out[(rowBlock + m0 + r) * 256 + col] = (bf16_t)v;
            s1 += v;
            s2 += v * v;
        }
        atomicAdd(&sSum[col], s1);
        atomicAdd(&sSqr[col], s2);
    }
    __syncthreads();
    atomicAdd(&a.sums[tid],  sSum[tid]);
    atomicAdd(&a.sumsq[tid], sSqr[tid]);
}

// ---------------------------------------------------------------- weight swizzle with BN-scale fold (f32 -> bf16)

__global__ __launch_bounds__(256) void nri_swizzle_w(const float* __restrict__ W,
                                                     const float* __restrict__ sc0,
                                                     const float* __restrict__ sc1,
                                                     const float* __restrict__ sc2,
                                                     bf16_t* __restrict__ out) {
    const int idx = blockIdx.x * 256 + threadIdx.x;     // covers K*256
    const int j  = idx & 15;
    const int l  = (idx >> 4) & 31;
    const int nt = (idx >> 9) & 15;
    const int kb = idx >> 13;
    const int k  = kb * 32 + ((l >> 4) << 3) + (j < 8 ? j : j + 8);
    const int n  = nt * 16 + (l & 15);
    const float* sc = (k < 256) ? sc0 : ((k < 512) ? sc1 : sc2);
    const float m = sc ? sc[k & 255] : 1.0f;
    out[idx] = (bf16_t)(W[k * 256 + n] * m);
}

// bias fold: b'[n] = b[n] + sum_k shift(k) * W[k,n]
__global__ __launch_bounds__(256) void nri_fold_bias(const float* __restrict__ W,
                                                     const float* __restrict__ sh0,
                                                     const float* __restrict__ sh1,
                                                     const float* __restrict__ sh2,
                                                     const float* __restrict__ b,
                                                     float* __restrict__ bOut,
                                                     int K) {
    const int n = threadIdx.x;
    float s = b[n];
    for (int k = 0; k < K; ++k) {
        const float* sh = (k < 256) ? sh0 : ((k < 512) ? sh1 : sh2);
        if (sh) s += sh[k & 255] * W[k * 256 + n];
    }
    bOut[n] = s;
}

// ---------------------------------------------------------------- BN finalize

__global__ __launch_bounds__(256) void nri_bn_finalize(const float* __restrict__ sums,
                                                       const float* __restrict__ sumsq,
                                                       const float* __restrict__ gamma,
                                                       const float* __restrict__ beta,
                                                       float* __restrict__ scale,
                                                       float* __restrict__ shift,
                                                       float invRows) {
    const int c = threadIdx.x;
    const float mean = sums[c] * invRows;
    const float var  = sumsq[c] * invRows - mean * mean;
    const float sc   = gamma[c] * __frsqrt_rn(var + 1e-5f);
    scale[c] = sc;
    shift[c] = beta[c] - mean * sc;
}

// ---------------------------------------------------------------- f32 -> bf16 convert

__global__ __launch_bounds__(256) void nri_cvt_bf16(const float* __restrict__ in,
                                                    bf16_t* __restrict__ out) {
    const int idx = blockIdx.x * 256 + threadIdx.x;
    out[idx] = (bf16_t)in[idx];
}

// ---------------------------------------------------------------- edge -> node aggregation (mlp2 BN folded, bf16 out)

__global__ __launch_bounds__(256) void nri_edge2node(const bf16_t* __restrict__ x2,
                                                     const float* __restrict__ scale2,
                                                     const float* __restrict__ shift2,
                                                     bf16_t* __restrict__ incoming) {
    const int blk = blockIdx.x;          // b * 64 + n
    const int b = blk >> 6;
    const int n = blk & 63;
    const int c = threadIdx.x;
    float s = 0.0f;
    for (int i = 0; i < NATOMS; ++i) {
        if (i == n) continue;
        const int e = i * 63 + (n - (n > i ? 1 : 0));
        s += (float)x2[(b * EDGES + e) * 256 + c];
    }
    const float v = (scale2[c] * s + 63.0f * shift2[c]) * (1.0f / 64.0f);
    incoming[(b * NATOMS + n) * 256 + c] = (bf16_t)v;
}

// ---------------------------------------------------------------- fc_out (256 -> 2) with mlp4 BN folded

__global__ __launch_bounds__(256) void nri_fcout(const bf16_t* __restrict__ h4,
                                                 const float* __restrict__ scale4,
                                                 const float* __restrict__ shift4,
                                                 const float* __restrict__ Wout,
                                                 const float* __restrict__ bout,
                                                 float* __restrict__ out) {
    const int wave = threadIdx.x >> 5;
    const int lane = threadIdx.x & 31;
    const int row  = blockIdx.x * 8 + wave;
    const bf16_t* hp = h4 + (size_t)row * 256;
    float p0 = 0.f, p1 = 0.f;
#pragma unroll
    for (int c0 = 0; c0 < 8; ++c0) {
        const int c = c0 * 32 + lane;
        const float v = (float)hp[c] * scale4[c] + shift4[c];
        p0 += v * Wout[c * 2 + 0];
        p1 += v * Wout[c * 2 + 1];
    }
#pragma unroll
    for (int off = 16; off > 0; off >>= 1) {
        p0 += __shfl_xor(p0, off, 32);
        p1 += __shfl_xor(p1, off, 32);
    }
    if (lane == 0) {
        out[row * 2 + 0] = p0 + bout[0];
        out[row * 2 + 1] = p1 + bout[1];
    }
}

// ---------------------------------------------------------------- launch

extern "C" void kernel_launch(void* const* d_in, const int* in_sizes, int n_in,
                              void* d_out, int out_size, void* d_ws, size_t ws_size,
                              hipStream_t stream) {
    (void)in_sizes; (void)n_in; (void)out_size; (void)ws_size;

    const float* inputs = (const float*)d_in[0];
    // d_in[1] = rel_rec, d_in[2] = rel_send : replaced by analytic index math

    struct MlpP { const float *W1, *b1, *W2, *b2, *gamma, *beta; };
    MlpP mp[4];
    int p = 3;
    for (int k = 0; k < 4; ++k) {
        mp[k].W1    = (const float*)d_in[p + 0];
        mp[k].b1    = (const float*)d_in[p + 1];
        mp[k].W2    = (const float*)d_in[p + 2];
        mp[k].b2    = (const float*)d_in[p + 3];
        mp[k].gamma = (const float*)d_in[p + 4];
        mp[k].beta  = (const float*)d_in[p + 5];
        p += 6;
    }
    const float* foW = (const float*)d_in[27];
    const float* fob = (const float*)d_in[28];

    // workspace carve-up
    char* base = (char*)d_ws;
    size_t off = 0;
    auto alloc = [&](size_t bytes) -> char* {
        char* r = base + off;
        off += (bytes + 255) & ~(size_t)255;
        return r;
    };

    const int K1[4] = {256, 512, 256, 768};
    bf16_t *W1s[4], *W2s[4];
    for (int k = 0; k < 4; ++k) {
        W1s[k] = (bf16_t*)alloc((size_t)K1[k] * 256 * sizeof(bf16_t));
        W2s[k] = (bf16_t*)alloc((size_t)256 * 256 * sizeof(bf16_t));
    }
    float* statsBase = (float*)alloc(4 * 4 * 256 * sizeof(float));
    float *sums[4], *sumsq[4], *scale[4], *shift[4];
    for (int k = 0; k < 4; ++k) {
        sums[k]  = statsBase + k * 1024 + 0;
        sumsq[k] = statsBase + k * 1024 + 256;
        scale[k] = statsBase + k * 1024 + 512;
        shift[k] = statsBase + k * 1024 + 768;
    }
    float*  biasF    = (float*) alloc(4 * 256 * sizeof(float));     // folded b1 per mlp
    bf16_t* xin      = (bf16_t*)alloc((size_t)NROWS * 256 * sizeof(bf16_t));
    bf16_t* h1       = (bf16_t*)alloc((size_t)NROWS * 256 * sizeof(bf16_t));
    bf16_t* h3       = (bf16_t*)alloc((size_t)NROWS * 256 * sizeof(bf16_t));
    bf16_t* incoming = (bf16_t*)alloc((size_t)NROWS * 256 * sizeof(bf16_t));
    bf16_t* x2       = (bf16_t*)alloc((size_t)EROWS * 256 * sizeof(bf16_t)); // reused as h4

    hipMemsetAsync(statsBase, 0, 4 * 4 * 256 * sizeof(float), stream);

    // input f32 -> bf16 ; layer-2 weights never see input affine: swizzle once
    nri_cvt_bf16<<<NROWS, 256, 0, stream>>>(inputs, xin);
    for (int k = 0; k < 4; ++k)
        nri_swizzle_w<<<256, 256, 0, stream>>>(mp[k].W2, nullptr, nullptr, nullptr, W2s[k]);

    GemmArgs a;

    // ---- mlp1
    nri_swizzle_w<<<K1[0], 256, 0, stream>>>(mp[0].W1, nullptr, nullptr, nullptr, W1s[0]);
    nri_fold_bias<<<1, 256, 0, stream>>>(mp[0].W1, nullptr, nullptr, nullptr, mp[0].b1, biasF + 0, K1[0]);
    a.seg[0] = xin; a.seg[1] = nullptr; a.seg[2] = nullptr;
    a.rowMode[0] = 0; a.rowMode[1] = 0; a.rowMode[2] = 0;
    a.W1 = W1s[0]; a.b1 = biasF + 0; a.W2 = W2s[0]; a.b2 = mp[0].b2;
    a.out = h1; a.sums = sums[0]; a.sumsq = sumsq[0]; a.nKB = 8;
    nri_mlp2layer<<<NROWS / 64, 256, 0, stream>>>(a);
    nri_bn_finalize<<<1, 256, 0, stream>>>(sums[0], sumsq[0], mp[0].gamma, mp[0].beta,
                                           scale[0], shift[0], 1.0f / NROWS);

    // ---- mlp2: [send(h1)|rec(h1)], BN1 folded into W1/b1
    nri_swizzle_w<<<K1[1], 256, 0, stream>>>(mp[1].W1, scale[0], scale[0], nullptr, W1s[1]);
    nri_fold_bias<<<1, 256, 0, stream>>>(mp[1].W1, shift[0], shift[0], nullptr, mp[1].b1, biasF + 256, K1[1]);
    a.seg[0] = h1; a.seg[1] = h1; a.seg[2] = nullptr;
    a.rowMode[0] = 1; a.rowMode[1] = 2; a.rowMode[2] = 0;
    a.W1 = W1s[1]; a.b1 = biasF + 256; a.W2 = W2s[1]; a.b2 = mp[1].b2;
    a.out = x2; a.sums = sums[1]; a.sumsq = sumsq[1]; a.nKB = 16;
    nri_mlp2layer<<<EROWS / 64, 256, 0, stream>>>(a);
    nri_bn_finalize<<<1, 256, 0, stream>>>(sums[1], sumsq[1], mp[1].gamma, mp[1].beta,
                                           scale[1], shift[1], 1.0f / EROWS);

    // ---- edge -> node (BN2 applied here), bf16 out
    nri_edge2node<<<BATCH * NATOMS, 256, 0, stream>>>(x2, scale[1], shift[1], incoming);

    // ---- mlp3
    nri_swizzle_w<<<K1[2], 256, 0, stream>>>(mp[2].W1, nullptr, nullptr, nullptr, W1s[2]);
    nri_fold_bias<<<1, 256, 0, stream>>>(mp[2].W1, nullptr, nullptr, nullptr, mp[2].b1, biasF + 512, K1[2]);
    a.seg[0] = incoming; a.seg[1] = nullptr; a.seg[2] = nullptr;
    a.rowMode[0] = 0; a.rowMode[1] = 0; a.rowMode[2] = 0;
    a.W1 = W1s[2]; a.b1 = biasF + 512; a.W2 = W2s[2]; a.b2 = mp[2].b2;
    a.out = h3; a.sums = sums[2]; a.sumsq = sumsq[2]; a.nKB = 8;
    nri_mlp2layer<<<NROWS / 64, 256, 0, stream>>>(a);
    nri_bn_finalize<<<1, 256, 0, stream>>>(sums[2], sumsq[2], mp[2].gamma, mp[2].beta,
                                           scale[2], shift[2], 1.0f / NROWS);

    // ---- mlp4: [send(h3)|rec(h3)|x_skip(x2)]; BN3 on segs 0/1, BN2 on seg 2; out aliases x2
    nri_swizzle_w<<<K1[3], 256, 0, stream>>>(mp[3].W1, scale[2], scale[2], scale[1], W1s[3]);
    nri_fold_bias<<<1, 256, 0, stream>>>(mp[3].W1, shift[2], shift[2], shift[1], mp[3].b1, biasF + 768, K1[3]);
    a.seg[0] = h3; a.seg[1] = h3; a.seg[2] = x2;
    a.rowMode[0] = 1; a.rowMode[1] = 2; a.rowMode[2] = 0;
    a.W1 = W1s[3]; a.b1 = biasF + 768; a.W2 = W2s[3]; a.b2 = mp[3].b2;
    a.out = x2; a.sums = sums[3]; a.sumsq = sumsq[3]; a.nKB = 24;
    nri_mlp2layer<<<EROWS / 64, 256, 0, stream>>>(a);
    nri_bn_finalize<<<1, 256, 0, stream>>>(sums[3], sumsq[3], mp[3].gamma, mp[3].beta,
                                           scale[3], shift[3], 1.0f / EROWS);

    // ---- fc_out with mlp4 BN folded
    nri_fcout<<<EROWS / 8, 256, 0, stream>>>(x2, scale[3], shift[3], foW, fob, (float*)d_out);
}
